// GCNModel_42374147342661
// MI455X (gfx1250) — matrix operand order repaired and checked by reading, not compiled, not run on hardware
//
#include <hip/hip_runtime.h>
#include <math.h>

// ---------------- problem constants (match reference) ----------------
#define NNODES 100000
#define FIN    256
#define HID    32
#define NCLS   10

typedef __attribute__((ext_vector_type(16))) _Float16 v16h;
typedef __attribute__((ext_vector_type(8)))  float    v8f;

// ---------------------------------------------------------------------
// Pass 1: deg = 1.0 (self-loop pre-counted)
__global__ void k_init_deg(float* __restrict__ deg) {
    int i = blockIdx.x * blockDim.x + threadIdx.x;
    if (i < NNODES) deg[i] = 1.0f;
}

// Pass 2: deg[col[e]] += 1 over the E raw edges
__global__ void k_deg(const int* __restrict__ col, float* __restrict__ deg, int E) {
    int e = blockIdx.x * blockDim.x + threadIdx.x;
    if (e < E) atomicAdd(&deg[col[e]], 1.0f);
}

// Pass 3: deg -> deg^{-1/2} in place
__global__ void k_dinv(float* __restrict__ deg) {
    int i = blockIdx.x * blockDim.x + threadIdx.x;
    if (i < NNODES) deg[i] = rsqrtf(deg[i]);
}

// ---------------------------------------------------------------------
// Pass 4: h' = (feature @ W1) * dinv[row]   via v_wmma_f32_16x16x32_f16
// Each wave computes a 16-row x 32-col tile; 8 waves/block -> 128 rows/block.
// W1 is pre-swizzled in LDS into the per-lane B-fragment layout so every
// fragment is one 32-byte contiguous LDS read (2x ds_load_b128).
// s is initialized to h' (this is the self-loop contribution).
__global__ void k_gemm1(const float* __restrict__ feat,
                        const float* __restrict__ W1,
                        const float* __restrict__ dinv,
                        float* __restrict__ hprime,
                        float* __restrict__ s)
{
    // [ktile t:8][ntile f:2][lane:32][j:16] f16  = 16 KB
    __shared__ alignas(32) _Float16 w1s[8 * 2 * 32 * 16];
    for (int i = threadIdx.x; i < 8 * 2 * 32 * 16; i += blockDim.x) {
        int j = i & 15;
        int l = (i >> 4) & 31;
        int f = (i >> 9) & 1;
        int t = i >> 10;
        int k = t * 32 + ((l >> 4) * 8) + (j & 7) + ((j >> 3) * 16);     // B layout: K = 8*(lane>=16)+(j&7)+16*(j>=8)
        int c = f * 16 + (l & 15);                //           col = lane&15
        w1s[i] = (_Float16)W1[k * HID + c];
    }
    __syncthreads();

    const int lane  = threadIdx.x & 31;
    const int wave  = threadIdx.x >> 5;
    const int m0    = (blockIdx.x * 8 + wave) * 16;   // wave's first row
    const int mrow  = lane & 15;
    const int khalf = (lane >> 4) * 8;                // A layout: 0 / 8
    const int n     = lane & 15;                      // D layout: column

    int r = m0 + mrow; if (r >= NNODES) r = NNODES - 1;   // clamp (EXEC stays full)
    const float* fp = feat + (size_t)r * FIN;

    v8f c0 = {}; v8f c1 = {};
    #pragma unroll 2
    for (int t = 0; t < 8; ++t) {
        const int kk = t * 32;
        // prefetch next K-tile of this row into WGP-level cache
        if (t < 7) __builtin_prefetch(fp + kk + 32, 0, 3);
        // A fragment: two runs of 8 contiguous f32 -> 4x global_load_b128
        const float4* fa = (const float4*)(fp + kk + khalf);
        const float4* fb = (const float4*)(fp + kk + 16 + khalf);
        float4 x0 = fa[0], x1 = fa[1];
        float4 y0 = fb[0], y1 = fb[1];
        v16h a;
        a[0]  = (_Float16)x0.x; a[1]  = (_Float16)x0.y;
        a[2]  = (_Float16)x0.z; a[3]  = (_Float16)x0.w;
        a[4]  = (_Float16)x1.x; a[5]  = (_Float16)x1.y;
        a[6]  = (_Float16)x1.z; a[7]  = (_Float16)x1.w;
        a[8]  = (_Float16)y0.x; a[9]  = (_Float16)y0.y;
        a[10] = (_Float16)y0.z; a[11] = (_Float16)y0.w;
        a[12] = (_Float16)y1.x; a[13] = (_Float16)y1.y;
        a[14] = (_Float16)y1.z; a[15] = (_Float16)y1.w;
        // B fragments: 32 contiguous bytes each (2x ds_load_b128)
        v16h b0 = *(const v16h*)&w1s[((t * 2 + 0) * 32 + lane) * 16];
        v16h b1 = *(const v16h*)&w1s[((t * 2 + 1) * 32 + lane) * 16];
        c0 = __builtin_amdgcn_wmma_f32_16x16x32_f16(false, a, false, b0,
                                                    (short)0, c0, false, false);
        c1 = __builtin_amdgcn_wmma_f32_16x16x32_f16(false, a, false, b1,
                                                    (short)0, c1, false, false);
    }

    // D layout: lane holds col n, VGPR v holds row m = v + 8*(lane>=16)
    const int mb = (lane >> 4) * 8;
    #pragma unroll
    for (int v = 0; v < 8; ++v) {
        int m = m0 + mb + v;
        if (m < NNODES) {
            float dsc = dinv[m];
            float x0 = c0[v] * dsc, x1 = c1[v] * dsc;
            size_t o = (size_t)m * HID;
            hprime[o + n]      = x0;  hprime[o + 16 + n] = x1;
            s[o + n]           = x0;  s[o + 16 + n]      = x1;   // self-loop init
        }
    }
}

// ---------------------------------------------------------------------
// Pass 5: s[col] += h'[row]  (32 consecutive lanes = one edge, coalesced)
__global__ void k_scatter(const int* __restrict__ row,
                          const int* __restrict__ col,
                          const float* __restrict__ hprime,
                          float* __restrict__ s, int E)
{
    long long t = (long long)blockIdx.x * blockDim.x + threadIdx.x;
    long long e = t >> 5;
    int f = (int)(t & 31);
    if (e >= E) return;
    int rs = row[e];
    int cd = col[e];
    atomicAdd(&s[(size_t)cd * HID + f], hprime[(size_t)rs * HID + f]);
}

// ---------------------------------------------------------------------
// Pass 6: act = relu(dinv*s + b1); logits = act @ W2pad + b2pad; log_softmax
// One WMMA (16 nodes) per wave; log-softmax via width-16 shuffle reductions
// (matches D layout: row m = v + 8*(lane>=16) lives across a 16-lane half).
__global__ void k_out(const float* __restrict__ s,
                      const float* __restrict__ dinv,
                      const float* __restrict__ b1,
                      const float* __restrict__ W2,
                      const float* __restrict__ b2,
                      float* __restrict__ out)
{
    // W2 padded to 32x16, pre-swizzled into per-lane fragment layout: 1 KB
    __shared__ alignas(32) _Float16 w2s[32 * 16];
    for (int i = threadIdx.x; i < 32 * 16; i += blockDim.x) {
        int j = i & 15;
        int l = i >> 4;
        int k = ((l >> 4) * 8) + (j & 7) + ((j >> 3) * 16);
        int nn = l & 15;
        w2s[i] = (nn < NCLS) ? (_Float16)W2[k * NCLS + nn] : (_Float16)0.0f;
    }
    __syncthreads();

    const int lane  = threadIdx.x & 31;
    const int wave  = threadIdx.x >> 5;
    const int m0    = (blockIdx.x * 8 + wave) * 16;
    const int mrow  = lane & 15;
    const int khalf = (lane >> 4) * 8;
    const int n     = lane & 15;

    int r = m0 + mrow; if (r >= NNODES) r = NNODES - 1;   // clamp, EXEC stays full
    float dsc = dinv[r];
    const float* sp = s + (size_t)r * HID;

    const float4* fa = (const float4*)(sp + khalf);
    const float4* fb = (const float4*)(sp + 16 + khalf);
    float4 x0 = fa[0], x1 = fa[1];
    float4 y0 = fb[0], y1 = fb[1];
    float xs[16] = { x0.x, x0.y, x0.z, x0.w, x1.x, x1.y, x1.z, x1.w,
                     y0.x, y0.y, y0.z, y0.w, y1.x, y1.y, y1.z, y1.w };
    v16h a;
    #pragma unroll
    for (int j = 0; j < 8; ++j) {
        int k = khalf + j;
        a[j] = (_Float16)fmaxf(fmaf(dsc, xs[j], b1[k]), 0.0f);
    }
    #pragma unroll
    for (int j = 0; j < 8; ++j) {
        int k = 16 + khalf + j;
        a[8 + j] = (_Float16)fmaxf(fmaf(dsc, xs[8 + j], b1[k]), 0.0f);
    }
    v16h b = *(const v16h*)&w2s[lane * 16];

    v8f c = {};
    c = __builtin_amdgcn_wmma_f32_16x16x32_f16(false, a, false, b,
                                               (short)0, c, false, false);

    // logits: lane col n; padded cols get -inf so they vanish in the reduction
    float bias = (n < NCLS) ? b2[n] : -INFINITY;
    const int mb = (lane >> 4) * 8;
    #pragma unroll
    for (int v = 0; v < 8; ++v) {
        float logit = c[v] + bias;
        float mx = logit;
        #pragma unroll
        for (int msk = 8; msk >= 1; msk >>= 1)
            mx = fmaxf(mx, __shfl_xor(mx, msk, 16));
        float ex = expf(logit - mx);                 // exp(-inf - mx) = 0
        float sum = ex;
        #pragma unroll
        for (int msk = 8; msk >= 1; msk >>= 1)
            sum += __shfl_xor(sum, msk, 16);
        float res = logit - (mx + logf(sum));
        int m = m0 + mb + v;
        if (m < NNODES && n < NCLS)
            out[(size_t)m * NCLS + n] = res;
    }
}

// ---------------------------------------------------------------------
extern "C" void kernel_launch(void* const* d_in, const int* in_sizes, int n_in,
                              void* d_out, int out_size, void* d_ws, size_t ws_size,
                              hipStream_t stream) {
    const float* feat = (const float*)d_in[0];
    const int*   eidx = (const int*)d_in[1];       // int32 (JAX default x64-off)
    const float* W1   = (const float*)d_in[2];
    const float* b1   = (const float*)d_in[3];
    const float* W2   = (const float*)d_in[4];
    const float* b2   = (const float*)d_in[5];
    float* out = (float*)d_out;

    const int E = in_sizes[1] / 2;                 // edge_index is [2, E]
    const int* erow = eidx;                        // sources
    const int* ecol = eidx + E;                    // targets

    // workspace: deg/dinv [N] | h' [N*32] | s [N*32]
    float* deg    = (float*)d_ws;
    float* hprime = deg + ((NNODES + 63) & ~63);
    float* sbuf   = hprime + (size_t)NNODES * HID;

    const int B = 256;
    k_init_deg<<<(NNODES + B - 1) / B, B, 0, stream>>>(deg);
    k_deg    <<<(E + B - 1) / B, B, 0, stream>>>(ecol, deg, E);
    k_dinv   <<<(NNODES + B - 1) / B, B, 0, stream>>>(deg);

    int gemm_blocks = (NNODES + 127) / 128;        // 128 rows per 8-wave block
    k_gemm1<<<gemm_blocks, B, 0, stream>>>(feat, W1, deg, hprime, sbuf);

    long long sc_threads = (long long)E * 32;
    int sc_blocks = (int)((sc_threads + B - 1) / B);
    k_scatter<<<sc_blocks, B, 0, stream>>>(erow, ecol, hprime, sbuf, E);

    k_out<<<gemm_blocks, B, 0, stream>>>(sbuf, deg, b1, W2, b2, out);
}
